// EquiDeformConv2d_3410204033218
// MI455X (gfx1250) — matrix operand order repaired and checked by reading, not compiled, not run on hardware
//
#include <hip/hip_runtime.h>

// ---------------- problem constants ----------------
#define Bn    8
#define CIN   128
#define COUT  128
#define Hn    64
#define Wn    64
#define HW    (Hn * Wn)        // 4096
#define KTAPS 9
#define KTOT  (CIN * KTAPS)    // 1152

// workspace layout (floats / ushorts)
#define PY_ELEMS    (Bn * KTAPS * HW)          // 294912 floats
#define WFRAG_ELEMS (KTAPS * 4 * 8 * 32 * 16)  // 147456 bf16 (ushort)

typedef __attribute__((ext_vector_type(16))) __bf16       v16bf;
typedef __attribute__((ext_vector_type(8)))  float        v8f;
typedef __attribute__((ext_vector_type(4)))  unsigned int u32x4;
typedef __attribute__((ext_vector_type(4)))  float        f32x4;

union FragU { u32x4 q[2]; v16bf v; };

static __device__ __forceinline__ unsigned int f32_to_bf16(float f) {
    union { float f; unsigned int u; } a; a.f = f;
    unsigned int u = a.u;
    // round-to-nearest-even
    return (u + 0x7FFFu + ((u >> 16) & 1u)) >> 16;
}
static __device__ __forceinline__ unsigned int pack_bf16x2(float lo, float hi) {
    return f32_to_bf16(lo) | (f32_to_bf16(hi) << 16);
}

// ============================================================
// Kernel 1: offset conv -> per-(b,k,h,w) sample positions py/px
// ============================================================
__global__ __launch_bounds__(256) void offset_kernel(
    const float* __restrict__ x, const float* __restrict__ w_off,
    const float* __restrict__ b_off,
    float* __restrict__ pyA, float* __restrict__ pxA)
{
    __shared__ float lw[4 * CIN * 9];   // only first 4 offset channels are used
    int tid = threadIdx.x;
    for (int i = tid; i < 4 * CIN * 9; i += 256) lw[i] = w_off[i];
    __syncthreads();

    int g  = blockIdx.x * 256 + tid;     // one thread per (b,h,w)
    int b  = g >> 12;
    int hw = g & (HW - 1);
    int h  = hw >> 6, w = hw & 63;
    const float* xb = x + (size_t)b * CIN * HW;

    // precompute clamped tap offsets + validity masks (branch-free halo)
    int   off9[9];
    float msk9[9];
    #pragma unroll
    for (int ky = 0; ky < 3; ++ky) {
        #pragma unroll
        for (int kx = 0; kx < 3; ++kx) {
            int yy = h + ky - 1, xx = w + kx - 1;
            bool ok = (yy >= 0) && (yy < Hn) && (xx >= 0) && (xx < Wn);
            int yc = min(max(yy, 0), Hn - 1);
            int xc = min(max(xx, 0), Wn - 1);
            off9[ky * 3 + kx] = yc * Wn + xc;
            msk9[ky * 3 + kx] = ok ? 1.f : 0.f;
        }
    }

    float a0 = b_off[0], a1 = b_off[1], a2 = b_off[2], a3 = b_off[3];

    for (int c = 0; c < CIN; ++c) {
        const float* xc = xb + c * HW;
        const float* lwc = lw + c * 9;
        #pragma unroll
        for (int t = 0; t < 9; ++t) {
            float vv = xc[off9[t]] * msk9[t];
            a0 += vv * lwc[0 * CIN * 9 + t];
            a1 += vv * lwc[1 * CIN * 9 + t];
            a2 += vv * lwc[2 * CIN * 9 + t];
            a3 += vv * lwc[3 * CIN * 9 + t];
        }
    }

    float shiftY = a0, shiftX = a1;
    float s1 = fmaxf(a2, 0.f) + 1.f;
    float s2 = fmaxf(a3, 0.f) + 1.f;

    const float BY[9] = {-1,-1,-1, 0,0,0, 1,1,1};
    const float BX[9] = {-1, 0, 1,-1,0,1,-1,0,1};
    const float CR[9] = { 1, 0, 1, 0,0,0, 1,0,1};
    const float EG[9] = { 0, 1, 0, 1,0,1, 0,1,0};

    #pragma unroll
    for (int k = 0; k < 9; ++k) {
        float sc = CR[k] * s1 + EG[k] * s2;
        float dy = BY[k] * (sc - 1.f) + shiftY;
        float dx = BX[k] * (sc - 1.f) + shiftX;
        size_t idx = (((size_t)b * 9 + k) * Hn + h) * Wn + w;
        pyA[idx] = (float)h + BY[k] + dy;
        pxA[idx] = (float)w + BX[k] + dx;
    }
}

// ============================================================
// Kernel 2: pre-swizzle weights into bf16 WMMA B-fragments.
// B tile is 32(K=c)x16(N=o); contiguous k order used here (A matches):
//   lane l: N = l&15, K = (l>>4)*16 + e,  e = element 0..15
// fragment id fi = (k*4 + cblock)*8 + couttile, 512 elems each.
// ============================================================
__global__ __launch_bounds__(256) void wprep_kernel(
    const float* __restrict__ w, unsigned short* __restrict__ wfrag)
{
    int g    = blockIdx.x * 256 + threadIdx.x;   // 147456 total
    int e    = g & 15;
    int lane = (g >> 4) & 31;
    int fi   = g >> 9;
    int ot   = fi & 7;
    int cbk  = fi >> 3;
    int cb   = cbk & 3;
    int k    = cbk >> 2;

    int K = ((lane >> 4) << 4) + e;          // 0..31
    int c = cb * 32 + K;
    int o = ot * 16 + (lane & 15);
    float val = w[(size_t)o * KTOT + c * 9 + k];
    wfrag[g] = (unsigned short)f32_to_bf16(val);
}

// ============================================================
// Kernel 3: deformable conv via WMMA bf16.
// Block = one (b,h) row: 64 pixels x 128 couts, 8 waves.
// wave wv: pt = wv&3 (pixel tile of 16), ch = wv>>2 (cout half of 64)
// A tile double-buffered -> 1 barrier per K-step.
// ============================================================
__global__ __launch_bounds__(256) void deform_wmma_kernel(
    const float* __restrict__ x,
    const float* __restrict__ pyA, const float* __restrict__ pxA,
    const unsigned short* __restrict__ wfrag,
    const float* __restrict__ bias, float* __restrict__ out)
{
    __shared__ float s_wy[64], s_wx[64];
    __shared__ int   s_y0[64], s_x0[64];
    __shared__ __align__(16) unsigned short s_fragA[2][4][32][16]; // double-buffered
    __shared__ float s_out[64][COUT + 4];   // padded: stride % 64 != 0

    int tid  = threadIdx.x;
    int blk  = blockIdx.x;           // 512 blocks
    int b    = blk >> 6;
    int h    = blk & 63;
    int lane = tid & 31;
    int wv   = tid >> 5;
    int pt   = wv & 3;
    int ch   = wv >> 2;

    // staging role (constant per thread)
    int p    = tid & 63;             // pixel in row
    int cgrp = tid >> 6;             // 0..3 -> 8 channels each
    int ptp  = p >> 4, lrow = p & 15;
    int srow = lrow + ((cgrp >> 1) << 4);  // lane row in fragment
    int scol = (cgrp & 1) << 3;            // element base (0 or 8)

    const float* xb    = x + (size_t)b * CIN * HW;
    const float* xgrp0 = xb + (size_t)(cgrp * 8) * HW;

    v8f acc[4] = {};   // 16 pixels x 64 couts per wave

    for (int k = 0; k < 9; ++k) {
        if (tid < 64) {
            size_t idx = (((size_t)b * 9 + k) * Hn + h) * Wn + tid;
            float py = pyA[idx], px = pxA[idx];
            float y0f = floorf(py), x0f = floorf(px);
            s_wy[tid] = py - y0f;  s_wx[tid] = px - x0f;
            s_y0[tid] = (int)y0f;  s_x0[tid] = (int)x0f;
        }
        __syncthreads();

        // ---- per-k bilinear setup, kept in registers across c-blocks ----
        int   y0 = s_y0[p],  x0 = s_x0[p];
        float wy = s_wy[p],  wx = s_wx[p];
        int y0c = min(max(y0, 0), Hn - 1), y1c = min(max(y0 + 1, 0), Hn - 1);
        int x0c = min(max(x0, 0), Wn - 1), x1c = min(max(x0 + 1, 0), Wn - 1);
        float my0 = (y0 >= 0 && y0 < Hn) ? 1.f : 0.f;
        float my1 = (y0 + 1 >= 0 && y0 + 1 < Hn) ? 1.f : 0.f;
        float mx0 = (x0 >= 0 && x0 < Wn) ? 1.f : 0.f;
        float mx1 = (x0 + 1 >= 0 && x0 + 1 < Wn) ? 1.f : 0.f;
        // validity folded into the 4 corner weights -> branch-free gathers
        float w00 = (1.f - wy) * (1.f - wx) * my0 * mx0;
        float w01 = (1.f - wy) * wx         * my0 * mx1;
        float w10 = wy * (1.f - wx)         * my1 * mx0;
        float w11 = wy * wx                 * my1 * mx1;
        int i00 = y0c * Wn + x0c, i01 = y0c * Wn + x1c;
        int i10 = y1c * Wn + x0c, i11 = y1c * Wn + x1c;

        for (int cb = 0; cb < 4; ++cb) {
            int step = k * 4 + cb;
            int buf  = step & 1;

            // hint-prefetch this wave's weight-fragment stream
            __builtin_prefetch(wfrag + (size_t)(step * 8 + ch * 4) * 512 + lane * 16, 0, 1);

            // ---- gather + bilinear, 8 channels, unconditional loads ----
            const float* xc = xgrp0 + (size_t)(cb * 32) * HW;
            float val[8];
            #pragma unroll
            for (int j = 0; j < 8; ++j) {
                const float* xcj = xc + (size_t)j * HW;
                val[j] = xcj[i00] * w00 + xcj[i01] * w01 +
                         xcj[i10] * w10 + xcj[i11] * w11;
            }
            // pack 8 consecutive fragment elements -> one ds_store_b128
            u32x4 pk;
            pk.x = pack_bf16x2(val[0], val[1]);
            pk.y = pack_bf16x2(val[2], val[3]);
            pk.z = pack_bf16x2(val[4], val[5]);
            pk.w = pack_bf16x2(val[6], val[7]);
            *reinterpret_cast<u32x4*>(&s_fragA[buf][ptp][srow][scol]) = pk;
            __syncthreads();

            // ---- load A fragment (2x ds_load_b128 per lane) ----
            FragU A;
            const u32x4* ap =
                reinterpret_cast<const u32x4*>(&s_fragA[buf][pt][lane][0]);
            A.q[0] = ap[0];
            A.q[1] = ap[1];

            // ---- 4 WMMAs: 16px x 64 couts ----
            #pragma unroll
            for (int ot = 0; ot < 4; ++ot) {
                int fi = step * 8 + (ch * 4 + ot);
                const u32x4* bp = reinterpret_cast<const u32x4*>(
                    wfrag + (size_t)fi * 512 + lane * 16);
                FragU Bf;
                Bf.q[0] = bp[0];
                Bf.q[1] = bp[1];
                acc[ot] = __builtin_amdgcn_wmma_f32_16x16x32_bf16(
                    false, A.v, false, Bf.v, (short)0, acc[ot], false, false);
            }
            // no second barrier: next step writes the other buffer, and each
            // wave's ds reads complete (dscnt wait) before it signals the
            // next step's barrier.
        }
    }
    __syncthreads();

    // ---- epilogue: D layout -> LDS transpose -> coalesced b128 stores ----
    // D: lane<16: N=lane, M=i ; lane>=16: N=lane-16, M=i+8
    int nlo = lane & 15;
    int prow0 = pt * 16 + ((lane >> 4) << 3);
    #pragma unroll
    for (int ot = 0; ot < 4; ++ot) {
        int o = (ch * 4 + ot) * 16 + nlo;
        #pragma unroll
        for (int i = 0; i < 8; ++i) {
            s_out[prow0 + i][o] = acc[ot][i];
        }
    }
    __syncthreads();

    {
        int o    = tid >> 1;
        int half = (tid & 1) * 32;
        float bo = bias[o];
        float* op = out + (((size_t)b * COUT + o) * Hn + h) * Wn + half;
        #pragma unroll
        for (int q = 0; q < 8; ++q) {
            f32x4 vv;
            vv.x = s_out[half + q * 4 + 0][o] + bo;
            vv.y = s_out[half + q * 4 + 1][o] + bo;
            vv.z = s_out[half + q * 4 + 2][o] + bo;
            vv.w = s_out[half + q * 4 + 3][o] + bo;
            reinterpret_cast<f32x4*>(op)[q] = vv;
        }
    }
}

// ============================================================
extern "C" void kernel_launch(void* const* d_in, const int* in_sizes, int n_in,
                              void* d_out, int out_size, void* d_ws, size_t ws_size,
                              hipStream_t stream) {
    const float* x     = (const float*)d_in[0];
    const float* w_off = (const float*)d_in[1];
    const float* b_off = (const float*)d_in[2];
    const float* w     = (const float*)d_in[3];
    const float* bias  = (const float*)d_in[4];
    float* out = (float*)d_out;

    float* pyA = (float*)d_ws;
    float* pxA = pyA + PY_ELEMS;
    unsigned short* wfrag = (unsigned short*)(pxA + PY_ELEMS);

    // 1) offsets -> sample positions   (32768 px / 256)
    offset_kernel<<<(Bn * HW) / 256, 256, 0, stream>>>(x, w_off, b_off, pyA, pxA);
    // 2) weight fragments in WMMA layout (147456 / 256)
    wprep_kernel<<<WFRAG_ELEMS / 256, 256, 0, stream>>>(w, wfrag);
    // 3) main deformable conv, one block per (b,h) row
    deform_wmma_kernel<<<Bn * Hn, 256, 0, stream>>>(x, pyA, pxA, wfrag, bias, out);
}
